// BranchAngularSeparationLoss_37074157699122
// MI455X (gfx1250) — compile-verified, matrix operand order checked by reading, not run on hardware
//
#include <hip/hip_runtime.h>
#include <hip/hip_bf16.h>
#include <math.h>

#define NSEG 256
#define DIM  64

typedef __attribute__((ext_vector_type(2)))  float    v2f;
typedef __attribute__((ext_vector_type(8)))  float    v8f;
typedef __attribute__((ext_vector_type(16))) _Float16 v16h;

// ---------------------------------------------------------------------------
// Kernel 0: zero the workspace accumulators (sums[256*64] + counts[256]).
// ---------------------------------------------------------------------------
__global__ void basl_zero(float* __restrict__ p, int n) {
  int i = blockIdx.x * blockDim.x + threadIdx.x;
  if (i < n) p[i] = 0.0f;
}

// ---------------------------------------------------------------------------
// Kernel 1: streaming pass over N rows. 16 lanes (half-wave) per row, float4
// per lane. Normalize row, scatter-add into LDS-privatized [256][64] bins via
// ds_add_f32, flush to global with atomics once per block.
// ---------------------------------------------------------------------------
__global__ void basl_accum(const float* __restrict__ emb,
                           const int*   __restrict__ midx,
                           const int*   __restrict__ seg,
                           float*       __restrict__ gsums,
                           float*       __restrict__ gcnt,
                           int n) {
  extern __shared__ float smem[];
  float* bins = smem;               // NSEG*DIM floats (64 KB)
  float* cnts = smem + NSEG * DIM;  // NSEG floats

  for (int i = threadIdx.x; i < NSEG * DIM; i += blockDim.x) bins[i] = 0.0f;
  for (int i = threadIdx.x; i < NSEG; i += blockDim.x) cnts[i] = 0.0f;
  __syncthreads();

  const int sub = threadIdx.x & 15;                                // lane in half-wave
  const int hw  = (blockIdx.x * blockDim.x + threadIdx.x) >> 4;    // half-wave id
  const int nhw = (gridDim.x * blockDim.x) >> 4;
  const float maxnorm = 1.0f - 1e-5f;

  for (int i = hw; i < n; i += nhw) {
    const int r = midx[i];                                         // gather index
    const float4 v = ((const float4*)(emb + (size_t)r * DIM))[sub];
    float ss = v.x * v.x + v.y * v.y + v.z * v.z + v.w * v.w;
    ss += __shfl_xor(ss, 1, 32);
    ss += __shfl_xor(ss, 2, 32);
    ss += __shfl_xor(ss, 4, 32);
    ss += __shfl_xor(ss, 8, 32);                                   // sum over 16 lanes
    const float nrm = sqrtf(ss);
    const float s   = (nrm > maxnorm) ? (maxnorm / fmaxf(nrm, 1e-8f)) : 1.0f;
    const float inv = s / fmaxf(nrm * s, 1e-8f);                   // direction scale

    const int b = seg[i];
    float* bin = bins + b * DIM + sub * 4;
    atomicAdd(bin + 0, v.x * inv);
    atomicAdd(bin + 1, v.y * inv);
    atomicAdd(bin + 2, v.z * inv);
    atomicAdd(bin + 3, v.w * inv);
    if (sub == 0) atomicAdd(&cnts[b], 1.0f);
  }
  __syncthreads();

  for (int i = threadIdx.x; i < NSEG * DIM; i += blockDim.x) {
    const float v = bins[i];
    if (v != 0.0f) atomicAdd(&gsums[i], v);
  }
  for (int i = threadIdx.x; i < NSEG; i += blockDim.x) {
    const float v = cnts[i];
    if (v != 0.0f) atomicAdd(&gcnt[i], v);
  }
}

// ---------------------------------------------------------------------------
// Kernel 2 (single block, 256 threads = 8 waves):
//   Phase A: centroids + cohesion (branch_cos = sums.centroid / count).
//   Phase B: 256x256 Gram matrix via WMMA, masked relu sum over upper triangle.
// ---------------------------------------------------------------------------
__global__ void basl_finalize(const float* __restrict__ gsums,
                              const float* __restrict__ gcnt,
                              float*       __restrict__ out) {
  extern __shared__ float smem[];
  float*    cent  = smem;               // NSEG*DIM f32 centroids (64 KB)
  float*    red   = smem + NSEG * DIM;  // [0]=cohesion sum, [1]=sep sum
  _Float16* centh = (_Float16*)(red + 2);

  if (threadIdx.x < 2) red[threadIdx.x] = 0.0f;
  __syncthreads();

  // ---- Phase A: one thread per branch -------------------------------------
  {
    const int b = threadIdx.x;
    const float cnt = fmaxf(gcnt[b], 1.0f);
    const float* sb = gsums + b * DIM;
    float nsq = 0.0f;
    #pragma unroll
    for (int k = 0; k < DIM; ++k) { const float m = sb[k] / cnt; nsq += m * m; }
    const float inv = 1.0f / fmaxf(sqrtf(nsq), 1e-12f);
    float dot = 0.0f;
    #pragma unroll
    for (int k = 0; k < DIM; ++k) {
      const float c = (sb[k] / cnt) * inv;
      cent[b * DIM + k] = c;
      dot += sb[k] * c;
    }
    atomicAdd(&red[0], 1.0f - dot / cnt);   // 1 - branch_cos
  }
  __syncthreads();

#if !__has_builtin(__builtin_amdgcn_wmma_f32_16x16x4_f32)
  for (int i = threadIdx.x; i < NSEG * DIM; i += blockDim.x)
    centh[i] = (_Float16)cent[i];
  __syncthreads();
#endif

  // ---- Phase B: WMMA over upper-triangular 16x16 tiles --------------------
  const int wave = threadIdx.x >> 5;
  const int lane = threadIdx.x & 31;
  const int lm   = lane & 15;
  const int lh   = lane >> 4;
  float sep = 0.0f;

  for (int t = wave; t < 136; t += 8) {       // 136 tiles with bj >= bi
    int bi = 0, rem = t;
    while (rem >= 16 - bi) { rem -= 16 - bi; ++bi; }
    const int bj = bi + rem;
    v8f c = {};

#if __has_builtin(__builtin_amdgcn_wmma_f32_16x16x4_f32)
    // f32 path: V_WMMA_F32_16X16X4_F32, 16 steps of K=4.
    const float* Ab = cent + (bi * 16 + lm) * DIM;
    const float* Bb = cent + (bj * 16 + lm) * DIM;
    #pragma unroll
    for (int k0 = 0; k0 < DIM; k0 += 4) {
      const int kb = k0 + 2 * lh;             // A: VGPR0=K0/K2, VGPR1=K1/K3
      v2f a, bm;
      a.x  = Ab[kb];  a.y  = Ab[kb + 1];
      bm.x = Bb[kb];  bm.y = Bb[kb + 1];      // B[k][n] = cent[n][k]
      c = __builtin_amdgcn_wmma_f32_16x16x4_f32(false, a, false, bm,
                                                (short)0, c, false, false);
    }
#else
    // f16 fallback: codegen-confirmed V_WMMA_F32_16X16X32_F16, 2 steps of K=32.
    const _Float16* Ab = centh + (bi * 16 + lm) * DIM;
    const _Float16* Bb = centh + (bj * 16 + lm) * DIM;
    #pragma unroll
    for (int k0 = 0; k0 < DIM; k0 += 32) {
      v16h a, bm;
      #pragma unroll
      for (int i = 0; i < 16; ++i) {
        // ISA 16-bit A layout: VGPR0..3 -> K = 8*lh + i, VGPR4..7 -> K = 16 + 8*lh + (i-8)
        const int k = k0 + ((i < 8) ? (8 * lh + i) : (16 + 8 * lh + (i - 8)));
        a[i]  = Ab[k];
        bm[i] = Bb[k];
      }
      c = __builtin_amdgcn_wmma_f32_16x16x32_f16(false, a, false, bm,
                                                 (short)0, c, false, false);
    }
#endif

    // C/D layout: vgpr v, lane L -> row m = 16*bi + v + 8*(L/16), col n = 16*bj + L%16
    #pragma unroll
    for (int v = 0; v < 8; ++v) {
      const int m  = bi * 16 + v + 8 * lh;
      const int nn = bj * 16 + lm;
      const float x = c[v] - 0.2f;            // COS_MARGIN
      if (nn > m && x > 0.0f) sep += x;       // strict upper triangle, relu
    }
  }

  sep += __shfl_xor(sep, 16, 32);
  sep += __shfl_xor(sep, 8, 32);
  sep += __shfl_xor(sep, 4, 32);
  sep += __shfl_xor(sep, 2, 32);
  sep += __shfl_xor(sep, 1, 32);
  if (lane == 0) atomicAdd(&red[1], sep);
  __syncthreads();

  if (threadIdx.x == 0)
    out[0] = red[0] * (1.0f / 256.0f) + red[1] * (1.0f / 32640.0f);
}

// ---------------------------------------------------------------------------
extern "C" void kernel_launch(void* const* d_in, const int* in_sizes, int n_in,
                              void* d_out, int out_size, void* d_ws, size_t ws_size,
                              hipStream_t stream) {
  const float* emb  = (const float*)d_in[0];   // embeddings [N,64]
  const int*   midx = (const int*)d_in[1];     // member_indices [N]
  const int*   seg  = (const int*)d_in[2];     // segment_ids [N]
  // d_in[3] = num_branches (fixed at 256 by the problem)

  const int N = in_sizes[0] / DIM;

  float* sums = (float*)d_ws;                  // [256*64]
  float* cnts = sums + NSEG * DIM;             // [256]
  float* out  = (float*)d_out;

  const int nacc = NSEG * DIM + NSEG;
  basl_zero<<<(nacc + 255) / 256, 256, 0, stream>>>(sums, nacc);

  const size_t lds1 = (size_t)(NSEG * DIM + NSEG) * sizeof(float);         // 66560 B
  basl_accum<<<512, 256, lds1, stream>>>(emb, midx, seg, sums, cnts, N);

  const size_t lds2 = (size_t)NSEG * DIM * sizeof(float) + 16 +
                      (size_t)NSEG * DIM * sizeof(_Float16);               // ~98 KB
  basl_finalize<<<1, 256, lds2, stream>>>(sums, cnts, out);
}
